// Attend_66116726554801
// MI455X (gfx1250) — matrix operand order repaired and checked by reading, not compiled, not run on hardware
//
#include <hip/hip_runtime.h>

#define SEQ   4096
#define BATCH 32
#define HID   512
#define ROWS  (SEQ*BATCH)
#define MB    128          // rows of x per block
#define NSTG  64           // W channels staged per pass
#define STAGES (HID/NSTG)  // 8 passes over output channels
#define CROWS 16           // rows per TDM chunk
#define NCHUNK (MB/CROWS)  // 8 chunks, ping-pong staged

typedef __attribute__((ext_vector_type(16))) __bf16 v16bf;
typedef __attribute__((ext_vector_type(8)))  __bf16 v8bf;
typedef __attribute__((ext_vector_type(8)))  float  v8f;
typedef __attribute__((ext_vector_type(4)))  unsigned tdm_g0_t;  // uint32x4
typedef __attribute__((ext_vector_type(8)))  int      tdm_g1_t;  // int32x8
typedef __attribute__((ext_vector_type(4)))  int      tdm_g2_t;  // int32x4

#if __has_builtin(__builtin_amdgcn_tensor_load_to_lds) && \
    __has_builtin(__builtin_amdgcn_s_wait_tensorcnt)
#define USE_TDM 1
#else
#define USE_TDM 0
#endif

__device__ __forceinline__ unsigned short f2bfu(float f) {
  unsigned u = __builtin_bit_cast(unsigned, f);
  unsigned r = u + 0x7FFFu + ((u >> 16) & 1u);   // round-to-nearest-even
  return (unsigned short)(r >> 16);
}

// packed 4x bf16 store (ds_store_b64)
__device__ __forceinline__ void st4bf(unsigned short* p, float4 v) {
  union { unsigned short u[4]; unsigned long long ll; } t;
  t.u[0] = f2bfu(v.x); t.u[1] = f2bfu(v.y);
  t.u[2] = f2bfu(v.z); t.u[3] = f2bfu(v.w);
  *(unsigned long long*)p = t.ll;
}

// CDNA5 hardware tanh transcendental (v_tanh_f32)
__device__ __forceinline__ float tanh_hw(float v) {
#if __has_builtin(__builtin_amdgcn_tanhf)
  return __builtin_amdgcn_tanhf(v);
#else
  float r;
  asm("v_tanh_f32 %0, %1" : "=v"(r) : "v"(v));
  return r;
#endif
}

// Build a 16xK bf16 WMMA fragment from two 16B LDS chunks
__device__ __forceinline__ v16bf ldfrag(const unsigned short* p0,
                                        const unsigned short* p1) {
  v8bf a0 = *(const v8bf*)p0;
  v8bf a1 = *(const v8bf*)p1;
  v16bf r;
  #pragma unroll
  for (int i = 0; i < 8; ++i) { r[i] = a0[i]; r[8 + i] = a1[i]; }
  return r;
}

#if USE_TDM
// Issue TDM load of CROWS x HID fp32 tile -> LDS (2-D descriptor, per ISA 8.3/8.4)
__device__ __forceinline__ void tdm_load_rows(const float* gsrc, unsigned lds_byte) {
  unsigned long long ga = (unsigned long long)(const void*)gsrc;
  tdm_g0_t g0;
  g0[0] = 1u;                                        // count=1, user descriptor
  g0[1] = lds_byte;                                  // lds_addr
  g0[2] = (unsigned)ga;                              // global_addr[31:0]
  g0[3] = (unsigned)((ga >> 32) & 0x01FFFFFFu) | 0x80000000u;  // addr[56:32] | type=2
  tdm_g1_t g1;
  g1[0] = 0x20000;                                   // data_size=2 (4B), mask=0
  g1[1] = (int)(HID << 16);                          // tensor_dim0 = 512 (lo16 @ bit48)
  g1[2] = (int)(CROWS << 16);                        // tensor_dim1 = 16  (lo16 @ bit80)
  g1[3] = (int)(HID << 16);                          // tile_dim0 = 512 (@bit112), dim1 hi=0
  g1[4] = CROWS;                                     // tile_dim1 = 16, tile_dim2 = 0
  g1[5] = HID;                                       // tensor_dim0_stride = 512
  g1[6] = 0;
  g1[7] = 0;
  tdm_g2_t gz = {0, 0, 0, 0};                        // 2-D: groups 2/3 unused
#if defined(__clang_major__) && (__clang_major__ >= 23)
  tdm_g1_t gz8 = {0, 0, 0, 0, 0, 0, 0, 0};
  __builtin_amdgcn_tensor_load_to_lds(g0, g1, gz, gz, gz8, 0);
#else
  __builtin_amdgcn_tensor_load_to_lds(g0, g1, gz, gz, 0);
#endif
}
#endif

// -------------------------------------------------------------------------
// Kernel 1: fused GEMM (x @ W^T + b) -> tanh -> dot(context) => similarity
// 8 waves = 4 M-positions x 2 N-positions; each wave owns a 32x32 output
// tile (4 WMMA accumulators) -> 2 ds_load_b128 per WMMA.
// x tile staged by the Tensor Data Mover into a ping-pong fp32 buffer and
// converted to the bf16 A-panel by the waves (DMA overlapped with convert).
// -------------------------------------------------------------------------
__global__ __launch_bounds__(256) void k_gemm_sim(
    const float* __restrict__ x, const float* __restrict__ W,
    const float* __restrict__ bias, const float* __restrict__ ctx,
    float* __restrict__ sim)
{
  __shared__ unsigned short xA[MB * HID];    // 128 KB: block's A rows, bf16
  __shared__ unsigned short bW[NSTG * HID];  //  64 KB: staged W rows [n][k], bf16
  __shared__ float simLds[MB];               // per-row similarity accum
#if USE_TDM
  __shared__ float xStage[2 * CROWS * HID];  //  64 KB: TDM ping-pong fp32 staging
#endif

  const int tid  = threadIdx.x;
  const int lane = tid & 31;
  const int wave = tid >> 5;
  const int mpos = wave & 3;                 // 4 M-positions (32 rows each)
  const int npos = wave >> 2;                // 2 N-positions (32 staged ch each)
  const int half = lane >> 4;
  const int l15  = lane & 15;
  const int r0   = blockIdx.x * MB;

  if (tid < MB) simLds[tid] = 0.0f;

#if USE_TDM
  // ---- TDM-staged x -> bf16 A-panel (ping-pong, DMA overlapped w/ convert) ----
  {
    const unsigned stage_base = (unsigned)(unsigned long long)(const void*)&xStage[0];
    if (wave == 0)
      tdm_load_rows(x + (size_t)r0 * HID, stage_base);            // chunk 0 -> buf 0
    for (int c = 0; c < NCHUNK; ++c) {
      if (wave == 0) {
        if (c + 1 < NCHUNK) {
          tdm_load_rows(x + (size_t)(r0 + (c + 1) * CROWS) * HID,
                        stage_base + ((unsigned)((c + 1) & 1)) * (CROWS * HID * 4));
          __builtin_amdgcn_s_wait_tensorcnt(1);   // chunk c landed, c+1 in flight
        } else {
          __builtin_amdgcn_s_wait_tensorcnt(0);   // last chunk landed
        }
      }
      __syncthreads();
      const float4* src = (const float4*)(xStage + (c & 1) * (CROWS * HID));
      unsigned short* dst = xA + c * (CROWS * HID);
      #pragma unroll
      for (int i = 0; i < CROWS * HID / 4 / 256; ++i) {
        int idx = i * 256 + tid;
        st4bf(dst + idx * 4, src[idx]);
      }
      __syncthreads();   // conversion done before TDM reuses this buffer
    }
  }
#else
  // ---- fallback: stage x rows -> bf16 LDS via VGPRs (coalesced float4) ----
  const float4* xg = (const float4*)(x + (size_t)r0 * HID);
  for (int i = tid; i < MB * HID / 4; i += 256) {
    float4 v = xg[i];
    st4bf(xA + i * 4, v);
  }
#endif

  float simAcc[16];
  #pragma unroll
  for (int j = 0; j < 16; ++j) simAcc[j] = 0.0f;

  for (int st = 0; st < STAGES; ++st) {
    __syncthreads();  // xA/simLds ready (first iter) / bW reusable (later)

    // ---- stage W rows ch0..ch0+63, bf16, n-major k-contiguous ----
    const float4* wg = (const float4*)(W + (size_t)st * NSTG * HID);
    for (int i = tid; i < NSTG * HID / 4; i += 256) {
      float4 v = wg[i];
      st4bf(bW + i * 4, v);
    }
    __syncthreads();

    // ---- 32x32 output tile per wave over K=512 ----
    v8f acc00 = {}, acc01 = {}, acc10 = {}, acc11 = {};
    #pragma unroll 2
    for (int kc = 0; kc < HID / 32; ++kc) {
      // A fragments: rows mpos*32 + {0,16} + l15
      int a0b = (mpos * 32 + l15) * HID + kc * 32 + (half << 3);
      int a1b = a0b + 16 * HID;
      v16bf A0 = ldfrag(xA + a0b, xA + a0b + 16);
      v16bf A1 = ldfrag(xA + a1b, xA + a1b + 16);
      // B fragments: staged-local channels npos*32 + {0,16} + l15
      int b0b = (npos * 32 + l15) * HID + kc * 32 + (half << 3);
      int b1b = b0b + 16 * HID;
      v16bf B0 = ldfrag(bW + b0b, bW + b0b + 16);
      v16bf B1 = ldfrag(bW + b1b, bW + b1b + 16);

      acc00 = __builtin_amdgcn_wmma_f32_16x16x32_bf16(false, A0, false, B0,
                                                      (short)0, acc00, false, false);
      acc01 = __builtin_amdgcn_wmma_f32_16x16x32_bf16(false, A0, false, B1,
                                                      (short)0, acc01, false, false);
      acc10 = __builtin_amdgcn_wmma_f32_16x16x32_bf16(false, A1, false, B0,
                                                      (short)0, acc10, false, false);
      acc11 = __builtin_amdgcn_wmma_f32_16x16x32_bf16(false, A1, false, B1,
                                                      (short)0, acc11, false, false);
    }

    // ---- epilogue: tanh(+bias) * context, accumulate per-row similarity ----
    int chA = st * NSTG + npos * 32 + l15;       // N-tile t=0 channel
    int chB = chA + 16;                          // N-tile t=1 channel
    float bA = bias[chA], cA = ctx[chA];
    float bB = bias[chB], cB = ctx[chB];
    #pragma unroll
    for (int j = 0; j < 8; ++j) {
      simAcc[j]     += tanh_hw(acc00[j] + bA) * cA;   // mt=0, t=0
      simAcc[j]     += tanh_hw(acc01[j] + bB) * cB;   // mt=0, t=1
      simAcc[8 + j] += tanh_hw(acc10[j] + bA) * cA;   // mt=1, t=0
      simAcc[8 + j] += tanh_hw(acc11[j] + bB) * cB;   // mt=1, t=1
    }
  }

  // ---- reduce over N within 16-lane groups, combine N-position waves ----
  #pragma unroll
  for (int mt = 0; mt < 2; ++mt) {
    #pragma unroll
    for (int j = 0; j < 8; ++j) {
      float v = simAcc[mt * 8 + j];
      v += __shfl_xor(v, 1, 16);
      v += __shfl_xor(v, 2, 16);
      v += __shfl_xor(v, 4, 16);
      v += __shfl_xor(v, 8, 16);
      if (l15 == 0)
        atomicAdd(&simLds[mpos * 32 + mt * 16 + j + (half << 3)], v);  // ds_add_f32
    }
  }
  __syncthreads();

  if (tid < MB) {
    int r = r0 + tid;                 // global row = s*BATCH + b
    int b = r & (BATCH - 1);
    int s = r >> 5;
    sim[b * SEQ + s] = simLds[tid];
  }
}

// -------------------------------------------------------------------------
// Kernel 2: in-place softmax over seq (4096) per batch row
// -------------------------------------------------------------------------
__global__ __launch_bounds__(256) void k_softmax(float* __restrict__ sim)
{
  float* row = sim + blockIdx.x * SEQ;
  const int tid = threadIdx.x;
  __shared__ float redM[8];
  __shared__ float redS[8];

  float v[16];
  float m = -3.402823466e+38f;
  #pragma unroll
  for (int i = 0; i < 16; ++i) { v[i] = row[tid + i * 256]; m = fmaxf(m, v[i]); }
  #pragma unroll
  for (int o = 16; o >= 1; o >>= 1) m = fmaxf(m, __shfl_xor(m, o, 32));
  if ((tid & 31) == 0) redM[tid >> 5] = m;
  __syncthreads();
  float bm = redM[0];
  #pragma unroll
  for (int i = 1; i < 8; ++i) bm = fmaxf(bm, redM[i]);

  float s = 0.0f;
  #pragma unroll
  for (int i = 0; i < 16; ++i) { v[i] = __expf(v[i] - bm); s += v[i]; }
  #pragma unroll
  for (int o = 16; o >= 1; o >>= 1) s += __shfl_xor(s, o, 32);
  if ((tid & 31) == 0) redS[tid >> 5] = s;
  __syncthreads();
  float bs = 0.0f;
  #pragma unroll
  for (int i = 0; i < 8; ++i) bs += redS[i];
  float inv = 1.0f / bs;
  #pragma unroll
  for (int i = 0; i < 16; ++i) row[tid + i * 256] = v[i] * inv;
}

// -------------------------------------------------------------------------
// Kernel 3: zero-init output, then probability-weighted pooling over seq
// -------------------------------------------------------------------------
__global__ void k_zero(float* __restrict__ out) {
  out[blockIdx.x * 256 + threadIdx.x] = 0.0f;
}

__global__ __launch_bounds__(256) void k_pool(
    const float* __restrict__ x, const float* __restrict__ probs,
    float* __restrict__ out)
{
  // grid = BATCH * 2 * (SEQ/256): b | h-chunk | seq-slice
  int blk = blockIdx.x;
  int b  = blk & (BATCH - 1);
  int hc = (blk >> 5) & 1;
  int sc = blk >> 6;
  int h  = hc * 256 + threadIdx.x;
  int s0 = sc * 256;
  float acc = 0.0f;
  #pragma unroll 4
  for (int i = 0; i < 256; ++i) {
    int s = s0 + i;
    acc += x[((size_t)s * BATCH + b) * HID + h] * probs[b * SEQ + s];
  }
  atomicAdd(&out[b * HID + h], acc);
}

// -------------------------------------------------------------------------
extern "C" void kernel_launch(void* const* d_in, const int* in_sizes, int n_in,
                              void* d_out, int out_size, void* d_ws, size_t ws_size,
                              hipStream_t stream)
{
  const float* x   = (const float*)d_in[0];
  const float* W   = (const float*)d_in[1];
  const float* b   = (const float*)d_in[2];
  const float* ctx = (const float*)d_in[3];
  float* out = (float*)d_out;
  float* sim = (float*)d_ws;               // BATCH*SEQ floats = 512 KB scratch

  k_gemm_sim<<<ROWS / MB, 256, 0, stream>>>(x, W, b, ctx, sim);
  k_softmax<<<BATCH, 256, 0, stream>>>(sim);
  k_zero<<<(BATCH * HID) / 256, 256, 0, stream>>>(out);
  k_pool<<<BATCH * 2 * (SEQ / 256), 256, 0, stream>>>(x, sim, out);
}
